// HardConstrainedMLP_unroll_4346506904071
// MI455X (gfx1250) — compile-verified, matrix operand order checked by reading, not run on hardware
//
#include <hip/hip_runtime.h>
#include <cstdint>
#include <cstddef>

// ---------------------------------------------------------------------------
// Types for CDNA5 WMMA (wave32, 16x16x32 bf16 -> f32)
// ---------------------------------------------------------------------------
typedef __attribute__((ext_vector_type(16))) __bf16 v16bf;
typedef __attribute__((ext_vector_type(8)))  __bf16 bf16x8;
typedef __attribute__((ext_vector_type(8)))  float  v8f;

__device__ __forceinline__ __bf16 f2bf(float f) {
    uint32_t u = __builtin_bit_cast(uint32_t, f);
    uint32_t r = u + 0x7FFFu + ((u >> 16) & 1u);   // round-to-nearest-even
    uint16_t h = (uint16_t)(r >> 16);
    return __builtin_bit_cast(__bf16, h);
}

__device__ __forceinline__ v8f wmma_bf16(v16bf a, v16bf b, v8f c) {
    return __builtin_amdgcn_wmma_f32_16x16x32_bf16(
        false, a, false, b, (short)0, c, false, false);
}

// ---------------------------------------------------------------------------
// gfx1250 async global->LDS copy (ASYNCcnt-tracked), ISA 15.18.3 opcode 98.
// One instruction moves 16B per active lane directly into LDS.
// ---------------------------------------------------------------------------
__device__ __forceinline__ void async_ld_b128(uint32_t lds_byte_addr, const void* gaddr) {
    asm volatile("global_load_async_to_lds_b128 %0, %1, off"
                 :: "v"(lds_byte_addr), "v"(gaddr) : "memory");
}
__device__ __forceinline__ void wait_async_le(int) = delete;
__device__ __forceinline__ void wait_async_le4() {
    asm volatile("s_wait_asynccnt 0x4" ::: "memory");
}
__device__ __forceinline__ void wait_async_0() {
    asm volatile("s_wait_asynccnt 0x0" ::: "memory");
}

// flat pointer -> LDS byte address (LDS aperture: addr[31:0] == LDS offset)
__device__ __forceinline__ uint32_t lds_addr(const void* p) {
    return (uint32_t)(uintptr_t)p;
}

// ---------------------------------------------------------------------------
// Fragment loaders. ISA 16-bit A/B layout:
//   lanes 0-15 : row=lane,    K {k0..k0+7, k0+16..k0+23}
//   lanes 16-31: row=lane-16, K {k0+8..k0+15, k0+24..k0+31}
// ---------------------------------------------------------------------------
__device__ __forceinline__ v16bf load_frag_bf16(const __bf16* __restrict__ p,
                                                int ld, int row_base, int k_base,
                                                int lane) {
    int lr = lane & 15, hi = lane >> 4;
    const __bf16* q = p + (size_t)(row_base + lr) * (size_t)ld + k_base + hi * 8;
    bf16x8 lo = *(const bf16x8*)(q);
    bf16x8 hb = *(const bf16x8*)(q + 16);
    v16bf out;
#pragma unroll
    for (int i = 0; i < 8; ++i) { out[i] = lo[i]; out[i + 8] = hb[i]; }
    return out;
}

// Fragment from an LDS-resident bf16 tile, row-major [rows][32] (ld == 32).
__device__ __forceinline__ v16bf load_frag_lds32(const __bf16* sh, int row_base, int lane) {
    int lr = lane & 15, hi = lane >> 4;
    const __bf16* q = sh + (row_base + lr) * 32 + hi * 8;
    bf16x8 lo = *(const bf16x8*)(q);
    bf16x8 hb = *(const bf16x8*)(q + 16);
    v16bf out;
#pragma unroll
    for (int i = 0; i < 8; ++i) { out[i] = lo[i]; out[i + 8] = hb[i]; }
    return out;
}

// Fragment from fp32 LDS with on-the-fly bf16 conversion (solver relayout).
__device__ __forceinline__ v16bf load_frag_f32lds(const float* __restrict__ sh,
                                                  int ld, int k_base, int lane) {
    int lr = lane & 15, hi = lane >> 4;
    const float* q = sh + lr * ld + k_base + hi * 8;
    v16bf out;
#pragma unroll
    for (int i = 0; i < 8; ++i) { out[i] = f2bf(q[i]); out[i + 8] = f2bf(q[i + 16]); }
    return out;
}

// ---------------------------------------------------------------------------
// Elementwise / transpose conversions
// ---------------------------------------------------------------------------
__global__ __launch_bounds__(256) void cvt_bf16_kernel(const float* __restrict__ in,
                                                       __bf16* __restrict__ out, int n) {
    int i = blockIdx.x * blockDim.x + threadIdx.x;
    if (i < n) out[i] = f2bf(in[i]);
}

__global__ __launch_bounds__(256) void transpose_cvt_kernel(const float* __restrict__ in,
                                                            __bf16* __restrict__ out,
                                                            int R, int C) {
    int i = blockIdx.x * blockDim.x + threadIdx.x;
    if (i < R * C) {
        int r = i / C, c = i - r * C;
        out[(size_t)c * R + r] = f2bf(in[i]);
    }
}

// ---------------------------------------------------------------------------
// bf16 WMMA GEMM, async-LDS double buffered.
//   Ab : [M,K] row-major bf16   BT : [N,K] row-major bf16 (pre-transposed)
// Block: 256 threads = 8 waves. Block tile 128(M) x 128(N).
// Wave (wm = w&3, wn = w>>2) computes 32(M) x 64(N): 8 accumulators.
// Per 32-deep K chunk: block stages A-tile[128x32] and B-tile[128x32] into
// LDS via global_load_async_to_lds_b128 (4 async instr / wave / chunk),
// prefetching chunk c+1 while computing chunk c.
// ---------------------------------------------------------------------------
__global__ __launch_bounds__(256) void gemm_bf16_kernel(
    const __bf16* __restrict__ Ab, const __bf16* __restrict__ BT,
    const float* __restrict__ bias,
    float* __restrict__ outf, __bf16* __restrict__ outb,
    int M, int N, int K, int relu) {

    __shared__ __bf16 Abuf[2][128 * 32];   // 8 KB per stage
    __shared__ __bf16 Bbuf[2][128 * 32];   // 8 KB per stage

    const int tid  = threadIdx.x;
    const int lane = tid & 31;
    const int wave = tid >> 5;
    const int lr = lane & 15, hi = lane >> 4;
    const int wm = wave & 3;           // 0..3 : 32-row strip
    const int wn = wave >> 2;          // 0..1 : 64-col strip
    const int bm = blockIdx.y * 128;
    const int bn = blockIdx.x * 128;

    // async-copy assignment: 2 x 16B segments per thread per tile
    // idx in [0,512): r = idx>>2 (tile row), seg = idx&3 (8 bf16 each)
    const int idx0 = tid * 2, idx1 = tid * 2 + 1;
    const int r0c = idx0 >> 2, s0c = idx0 & 3;
    const int r1c = idx1 >> 2, s1c = idx1 & 3;

    auto issue_chunk = [&](int c) {
        const int kc = c * 32;
        async_ld_b128(lds_addr(&Abuf[c & 1][r0c * 32 + s0c * 8]),
                      Ab + (size_t)(bm + r0c) * K + kc + s0c * 8);
        async_ld_b128(lds_addr(&Abuf[c & 1][r1c * 32 + s1c * 8]),
                      Ab + (size_t)(bm + r1c) * K + kc + s1c * 8);
        async_ld_b128(lds_addr(&Bbuf[c & 1][r0c * 32 + s0c * 8]),
                      BT + (size_t)(bn + r0c) * K + kc + s0c * 8);
        async_ld_b128(lds_addr(&Bbuf[c & 1][r1c * 32 + s1c * 8]),
                      BT + (size_t)(bn + r1c) * K + kc + s1c * 8);
    };

    v8f acc[2][4];
#pragma unroll
    for (int nt = 0; nt < 4; ++nt) {
        float bv = bias[bn + wn * 64 + nt * 16 + lr];
#pragma unroll
        for (int sub = 0; sub < 2; ++sub)
#pragma unroll
            for (int g = 0; g < 8; ++g) acc[sub][nt][g] = bv;
    }

    const int nc = K >> 5;
    issue_chunk(0);

    for (int c = 0; c < nc; ++c) {
        if (c + 1 < nc) { issue_chunk(c + 1); wait_async_le4(); }
        else            { wait_async_0(); }
        __syncthreads();                       // chunk c resident for all waves

        const __bf16* At = Abuf[c & 1];
        const __bf16* Bt = Bbuf[c & 1];
        v16bf a0 = load_frag_lds32(At, wm * 32, lane);
        v16bf a1 = load_frag_lds32(At, wm * 32 + 16, lane);
#pragma unroll
        for (int nt = 0; nt < 4; ++nt) {
            v16bf bf_ = load_frag_lds32(Bt, wn * 64 + nt * 16, lane);
            acc[0][nt] = wmma_bf16(a0, bf_, acc[0][nt]);
            acc[1][nt] = wmma_bf16(a1, bf_, acc[1][nt]);
        }
        __syncthreads();                       // safe to overwrite buf[c&1] later
    }

#pragma unroll
    for (int sub = 0; sub < 2; ++sub)
#pragma unroll
        for (int nt = 0; nt < 4; ++nt)
#pragma unroll
            for (int g = 0; g < 8; ++g) {
                float v = acc[sub][nt][g];
                if (relu) v = v > 0.0f ? v : 0.0f;
                int m = bm + wm * 32 + sub * 16 + g + hi * 8;
                int n = bn + wn * 64 + nt * 16 + lr;
                if (outb) outb[(size_t)m * N + n] = f2bf(v);
                else      outf[(size_t)m * N + n] = v;
            }
}

// ---------------------------------------------------------------------------
// Fused 100-iteration over-relaxed reflect/project solver.
// One wave per 16-row tile; z[16,256] fp32 lives in LDS.
//   r = z*A^T - b ; s = r*AAT_inv ; d = s*A ;
//   p = z - d ; q = clip(2p - z) ; z += OMEGA*(q - p)
// Final extra pass writes p_aff(z) = z - d.
// ---------------------------------------------------------------------------
__global__ __launch_bounds__(32) void solver_kernel(
    const float* __restrict__ y,      // [4096,256]
    const float* __restrict__ bmat,   // [4096,64]
    const __bf16* __restrict__ Abf,   // [64,256]  B-operand for z*A^T
    const __bf16* __restrict__ ATbf,  // [256,64]  B-operand for s*A
    const __bf16* __restrict__ invbf, // [64,64]   symmetric
    float* __restrict__ out) {        // [4096,256]

    __shared__ float zsh[16 * 256];
    __shared__ float tsh[16 * 64];

    const int lane = threadIdx.x;
    const int lr = lane & 15, hi = lane >> 4;
    const int r0 = blockIdx.x * 16;
    const float OMEGA = 1.7f;

    for (int i = lane; i < 16 * 256; i += 32)
        zsh[i] = y[(size_t)r0 * 256 + i];

    v8f rinit[4];
#pragma unroll
    for (int nt = 0; nt < 4; ++nt)
#pragma unroll
        for (int g = 0; g < 8; ++g)
            rinit[nt][g] = -bmat[(size_t)(r0 + g + hi * 8) * 64 + nt * 16 + lr];

    __syncthreads();

    for (int it = 0; it <= 100; ++it) {
        const bool final_pass = (it == 100);

        // ---- r = z @ A^T - b   (16 x 64, K=256) ----
        v8f racc[4];
#pragma unroll
        for (int nt = 0; nt < 4; ++nt) racc[nt] = rinit[nt];

        for (int ks = 0; ks < 8; ++ks) {
            v16bf za = load_frag_f32lds(zsh, 256, ks * 32, lane);
#pragma unroll
            for (int nt = 0; nt < 4; ++nt) {
                v16bf bf_ = load_frag_bf16(Abf, 256, nt * 16, ks * 32, lane);
                racc[nt] = wmma_bf16(za, bf_, racc[nt]);
            }
        }

        __syncthreads();
#pragma unroll
        for (int nt = 0; nt < 4; ++nt)
#pragma unroll
            for (int g = 0; g < 8; ++g)
                tsh[(g + hi * 8) * 64 + nt * 16 + lr] = racc[nt][g];
        __syncthreads();

        // ---- s = r @ AAT_inv   (16 x 64, K=64) ----
        v16bf ra0 = load_frag_f32lds(tsh, 64, 0, lane);
        v16bf ra1 = load_frag_f32lds(tsh, 64, 32, lane);
        v8f sacc[4];
#pragma unroll
        for (int nt = 0; nt < 4; ++nt) {
            v8f z8 = {};
            z8 = wmma_bf16(ra0, load_frag_bf16(invbf, 64, nt * 16, 0, lane), z8);
            z8 = wmma_bf16(ra1, load_frag_bf16(invbf, 64, nt * 16, 32, lane), z8);
            sacc[nt] = z8;
        }

        __syncthreads();
#pragma unroll
        for (int nt = 0; nt < 4; ++nt)
#pragma unroll
            for (int g = 0; g < 8; ++g)
                tsh[(g + hi * 8) * 64 + nt * 16 + lr] = sacc[nt][g];
        __syncthreads();

        v16bf sa0 = load_frag_f32lds(tsh, 64, 0, lane);
        v16bf sa1 = load_frag_f32lds(tsh, 64, 32, lane);

        // ---- d = s @ A  (16 x 256, K=64), fused z-update / output ----
#pragma unroll 4
        for (int nt16 = 0; nt16 < 16; ++nt16) {
            v8f dacc = {};
            dacc = wmma_bf16(sa0, load_frag_bf16(ATbf, 64, nt16 * 16, 0, lane), dacc);
            dacc = wmma_bf16(sa1, load_frag_bf16(ATbf, 64, nt16 * 16, 32, lane), dacc);
#pragma unroll
            for (int g = 0; g < 8; ++g) {
                int idx = (g + hi * 8) * 256 + nt16 * 16 + lr;
                float zo = zsh[idx];
                float dv = dacc[g];
                if (final_pass) {
                    out[(size_t)(r0 + g + hi * 8) * 256 + nt16 * 16 + lr] = zo - dv;
                } else {
                    float q = zo - 2.0f * dv;
                    q = fminf(1.0f, fmaxf(-1.0f, q));
                    zsh[idx] = zo + OMEGA * (q - (zo - dv));
                }
            }
        }
        __syncthreads();
    }
}

// ---------------------------------------------------------------------------
// Host-side launch
// ---------------------------------------------------------------------------
extern "C" void kernel_launch(void* const* d_in, const int* in_sizes, int n_in,
                              void* d_out, int out_size, void* d_ws, size_t ws_size,
                              hipStream_t stream) {
    (void)in_sizes; (void)n_in; (void)out_size; (void)ws_size;

    constexpr int B = 4096, DIN = 128, N = 256, M = 64, F = 512;

    const float* x    = (const float*)d_in[0];
    const float* bvec = (const float*)d_in[1];
    // d_in[2] = step (unused)
    const float* W0 = (const float*)d_in[3];
    const float* b0 = (const float*)d_in[4];
    const float* W1 = (const float*)d_in[5];
    const float* b1 = (const float*)d_in[6];
    const float* W2 = (const float*)d_in[7];
    const float* b2 = (const float*)d_in[8];
    const float* W3 = (const float*)d_in[9];
    const float* b3 = (const float*)d_in[10];
    const float* A    = (const float*)d_in[11];
    const float* AATi = (const float*)d_in[12];
    float* out = (float*)d_out;

    char* w = (char*)d_ws;
    auto alignup = [](size_t v) { return (v + 255) & ~(size_t)255; };
    size_t off = 0;
    auto take = [&](size_t bytes) { char* p = w + off; off += alignup(bytes); return p; };

    __bf16* xb   = (__bf16*)take((size_t)B * DIN * 2);
    __bf16* W0T  = (__bf16*)take((size_t)F * DIN * 2);
    __bf16* W1T  = (__bf16*)take((size_t)F * F * 2);
    __bf16* W2T  = (__bf16*)take((size_t)F * F * 2);
    __bf16* W3T  = (__bf16*)take((size_t)N * F * 2);
    __bf16* h0   = (__bf16*)take((size_t)B * F * 2);
    __bf16* h1   = (__bf16*)take((size_t)B * F * 2);
    __bf16* h2   = (__bf16*)take((size_t)B * F * 2);
    float*  ybuf = (float*) take((size_t)B * N * 4);
    __bf16* Abf  = (__bf16*)take((size_t)M * N * 2);
    __bf16* ATbf = (__bf16*)take((size_t)N * M * 2);
    __bf16* IVbf = (__bf16*)take((size_t)M * M * 2);

    auto grid1 = [](int n) { return dim3((n + 255) / 256); };

    cvt_bf16_kernel<<<grid1(B * DIN), 256, 0, stream>>>(x, xb, B * DIN);
    transpose_cvt_kernel<<<grid1(DIN * F), 256, 0, stream>>>(W0, W0T, DIN, F);
    transpose_cvt_kernel<<<grid1(F * F),   256, 0, stream>>>(W1, W1T, F, F);
    transpose_cvt_kernel<<<grid1(F * F),   256, 0, stream>>>(W2, W2T, F, F);
    transpose_cvt_kernel<<<grid1(F * N),   256, 0, stream>>>(W3, W3T, F, N);
    cvt_bf16_kernel<<<grid1(M * N), 256, 0, stream>>>(A, Abf, M * N);
    transpose_cvt_kernel<<<grid1(M * N), 256, 0, stream>>>(A, ATbf, M, N);
    cvt_bf16_kernel<<<grid1(M * M), 256, 0, stream>>>(AATi, IVbf, M * M);

    // MLP trunk (bias + ReLU fused), async-LDS double-buffered WMMA GEMMs
    gemm_bf16_kernel<<<dim3(F / 128, B / 128), 256, 0, stream>>>(
        xb, W0T, b0, nullptr, h0, B, F, DIN, 1);
    gemm_bf16_kernel<<<dim3(F / 128, B / 128), 256, 0, stream>>>(
        h0, W1T, b1, nullptr, h1, B, F, F, 1);
    gemm_bf16_kernel<<<dim3(F / 128, B / 128), 256, 0, stream>>>(
        h1, W2T, b2, nullptr, h2, B, F, F, 1);
    gemm_bf16_kernel<<<dim3(N / 128, B / 128), 256, 0, stream>>>(
        h2, W3T, b3, ybuf, nullptr, B, N, F, 0);

    // fused 100-iteration projection solver + final p_aff
    solver_kernel<<<dim3(B / 16), 32, 0, stream>>>(ybuf, bvec, Abf, ATbf, IVbf, out);
}
